// Sarn_78563541779011
// MI455X (gfx1250) — matrix operand order repaired and checked by reading, not compiled, not run on hardware
//
#include <hip/hip_runtime.h>

typedef __attribute__((ext_vector_type(16))) _Float16 v16h;
typedef __attribute__((ext_vector_type(8)))  _Float16 v8h;
typedef __attribute__((ext_vector_type(8)))  float    v8f;

union Frag16 { v16h v; v8h h[2]; };

// ---- problem constants -----------------------------------------------------
#define NB   128          // batch
#define CIN  1024
#define HH   14
#define WW   14
#define OO   196          // H*W objects
#define ROWS (NB*OO)      // 25088 (multiple of 16)
#define CF   128
#define TH   128
#define DQ   288          // 258 padded to mult of 32
#define DP   416          // 388 padded to mult of 32
#define HP   256
#define GT   256
#define FP   256
#define AA   32

// ---------------------------------------------------------------------------
// Prep kernels: fp32 -> f16 conversion / transpose / padding
// ---------------------------------------------------------------------------
__global__ void img_convert_kernel(const float* __restrict__ image,
                                   _Float16* __restrict__ imgT) {
  size_t idx = (size_t)blockIdx.x * blockDim.x + threadIdx.x;
  const size_t total = (size_t)NB * OO * CIN;
  if (idx >= total) return;
  int cin = (int)(idx % CIN);
  size_t np = idx / CIN;            // n*196 + p
  int p = (int)(np % OO);
  int n = (int)(np / OO);
  imgT[idx] = (_Float16)image[((size_t)n * CIN + cin) * OO + p];
}

__global__ void wt_convert_kernel(const float* __restrict__ conv_w,
                                  _Float16* __restrict__ wt9) {
  int idx = blockIdx.x * blockDim.x + threadIdx.x;   // 9*128*1024
  if (idx >= 9 * CF * CIN) return;
  int cin = idx % CIN;
  int rest = idx / CIN;
  int cf = rest % CF;
  int tap = rest / CF;
  wt9[idx] = (_Float16)conv_w[((size_t)cf * CIN + cin) * 9 + tap];
}

// WT[c][k] = (k < Kin) ? W[k][c] : 0  ; WT is [C][Kpad] f16
__global__ void w_transpose_kernel(const float* __restrict__ W,
                                   _Float16* __restrict__ WT,
                                   int Kin, int Kpad, int C) {
  int idx = blockIdx.x * blockDim.x + threadIdx.x;
  if (idx >= C * Kpad) return;
  int k = idx % Kpad;
  int c = idx / Kpad;
  WT[idx] = (k < Kin) ? (_Float16)W[(size_t)k * C + c] : (_Float16)0;
}

// qe layout: [n][p][DQ]: [0..127]=conv out (written by conv kernel),
// [128]=xc, [129]=yc, [130..257]=code, [258..287]=0
__global__ void assemble_qe_kernel(const float* __restrict__ code,
                                   _Float16* __restrict__ qe) {
  int idx = blockIdx.x;             // n*196+p
  int n = idx / OO, p = idx % OO;
  int oh = p / WW, ow = p % WW;
  _Float16* row = qe + (size_t)idx * DQ;
  int t = threadIdx.x;              // 160 threads
  if (t == 0) row[CF + 0] = (_Float16)(-1.0f + 2.0f * (float)oh / 13.0f);
  if (t == 1) row[CF + 1] = (_Float16)(-1.0f + 2.0f * (float)ow / 13.0f);
  if (t < TH) row[CF + 2 + t] = (_Float16)code[(size_t)n * TH + t];
  if (t >= TH && t < TH + (DQ - 258)) row[258 + (t - TH)] = (_Float16)0;
}

// ---------------------------------------------------------------------------
// Conv 3x3 as 9 shifted WMMA GEMMs with 4-way M register blocking.
// One wave computes a 64(cf) x 16(pixel) tile: the image B-fragment is loaded
// once per k-step and reused by 4 weight A-fragments.
// grid = (CF/64 = 2, 13 pixel-tiles, 128 batch), block = 32 (one wave)
// ---------------------------------------------------------------------------
__global__ void conv_wmma_kernel(const _Float16* __restrict__ imgT,  // [n][p][CIN]
                                 const _Float16* __restrict__ wt9,   // [tap][cf][CIN]
                                 const float* __restrict__ conv_b,
                                 _Float16* __restrict__ qe) {        // [n][p][DQ]
  const int lane = threadIdx.x;
  const int mt0 = blockIdx.x * 4;   // first cf tile of 4
  const int pt = blockIdx.y;        // pixel tile 0..12
  const int n  = blockIdx.z;
  const int half = lane >> 4;
  const int l15  = lane & 15;
  const int p  = pt * 16 + l15;     // pixel column for this lane (B side)
  const int oh = p / WW, ow = p % WW;

  v8f acc0 = {}, acc1 = {}, acc2 = {}, acc3 = {};
  const _Float16* abase = wt9 + (size_t)(mt0 * 16 + l15) * CIN + half * 16;
  const v8h zero = {};

  for (int tap = 0; tap < 9; ++tap) {
    const int dh = tap / 3 - 1, dw = tap % 3 - 1;
    const int sh = oh + dh, sw = ow + dw;
    const bool valid = (p < OO) && (sh >= 0) && (sh < HH) && (sw >= 0) && (sw < WW);
    const _Float16* bp = valid
        ? imgT + ((size_t)n * OO + sh * WW + sw) * CIN + half * 16
        : (const _Float16*)0;
    const _Float16* ap = abase + (size_t)tap * CF * CIN;
    for (int kk = 0; kk < CIN; kk += 32) {
      Frag16 b;
      if (valid) {
        b.h[0] = *(const v8h*)(bp);
        b.h[1] = *(const v8h*)(bp + 8);
        bp += 32;
      } else {
        b.h[0] = zero; b.h[1] = zero;
      }
      Frag16 a0, a1, a2, a3;
      a0.h[0] = *(const v8h*)(ap + 0 * 16 * CIN);
      a0.h[1] = *(const v8h*)(ap + 0 * 16 * CIN + 8);
      a1.h[0] = *(const v8h*)(ap + 1 * 16 * CIN);
      a1.h[1] = *(const v8h*)(ap + 1 * 16 * CIN + 8);
      a2.h[0] = *(const v8h*)(ap + 2 * 16 * CIN);
      a2.h[1] = *(const v8h*)(ap + 2 * 16 * CIN + 8);
      a3.h[0] = *(const v8h*)(ap + 3 * 16 * CIN);
      a3.h[1] = *(const v8h*)(ap + 3 * 16 * CIN + 8);
      acc0 = __builtin_amdgcn_wmma_f32_16x16x32_f16(false, a0.v, false, b.v,
                                                    (short)0, acc0, false, false);
      acc1 = __builtin_amdgcn_wmma_f32_16x16x32_f16(false, a1.v, false, b.v,
                                                    (short)0, acc1, false, false);
      acc2 = __builtin_amdgcn_wmma_f32_16x16x32_f16(false, a2.v, false, b.v,
                                                    (short)0, acc2, false, false);
      acc3 = __builtin_amdgcn_wmma_f32_16x16x32_f16(false, a3.v, false, b.v,
                                                    (short)0, acc3, false, false);
      ap += 32;
    }
  }

  // C layout: VGPR r holds (M = r + 8*half, N = lane&15)
  if (p < OO) {
    _Float16* out = qe + ((size_t)n * OO + p) * DQ;
    #pragma unroll
    for (int r = 0; r < 8; ++r) {
      int cf0 = (mt0 + 0) * 16 + r + half * 8;
      int cf1 = (mt0 + 1) * 16 + r + half * 8;
      int cf2 = (mt0 + 2) * 16 + r + half * 8;
      int cf3 = (mt0 + 3) * 16 + r + half * 8;
      out[cf0] = (_Float16)(acc0[r] + conv_b[cf0]);
      out[cf1] = (_Float16)(acc1[r] + conv_b[cf1]);
      out[cf2] = (_Float16)(acc2[r] + conv_b[cf2]);
      out[cf3] = (_Float16)(acc3[r] + conv_b[cf3]);
    }
  }
}

// ---------------------------------------------------------------------------
// Generic GEMM: C[rows][cols] = act(A[rows][K] * WT[cols][K]^T + bias)
// 4-way N register blocking: one wave computes a 16 x 64 tile; the A-fragment
// is loaded once per k-step and reused by 4 B-fragments.
// A row-major with lda == K (K multiple of 32), cols multiple of 64.
// grid = (rows/16, cols/64), block = 32
// ---------------------------------------------------------------------------
__global__ void gemm_wmma_kernel(const _Float16* __restrict__ A, int K,
                                 const _Float16* __restrict__ WT,
                                 const float* __restrict__ bias,
                                 _Float16* __restrict__ C, int cols,
                                 int do_relu) {
  const int lane = threadIdx.x;
  const int mt  = blockIdx.x;
  const int nt0 = blockIdx.y * 4;
  const int half = lane >> 4;
  const int l15  = lane & 15;

  const _Float16* ap = A  + (size_t)(mt * 16 + l15) * K + half * 16;
  const _Float16* bp = WT + (size_t)(nt0 * 16 + l15) * K + half * 16;
  const size_t bstride = (size_t)16 * K;   // between adjacent 16-col tiles

  v8f acc0 = {}, acc1 = {}, acc2 = {}, acc3 = {};
  for (int kk = 0; kk < K; kk += 32) {
    Frag16 a;
    a.h[0] = *(const v8h*)(ap);
    a.h[1] = *(const v8h*)(ap + 8);
    Frag16 b0, b1, b2, b3;
    b0.h[0] = *(const v8h*)(bp + 0 * bstride);
    b0.h[1] = *(const v8h*)(bp + 0 * bstride + 8);
    b1.h[0] = *(const v8h*)(bp + 1 * bstride);
    b1.h[1] = *(const v8h*)(bp + 1 * bstride + 8);
    b2.h[0] = *(const v8h*)(bp + 2 * bstride);
    b2.h[1] = *(const v8h*)(bp + 2 * bstride + 8);
    b3.h[0] = *(const v8h*)(bp + 3 * bstride);
    b3.h[1] = *(const v8h*)(bp + 3 * bstride + 8);
    acc0 = __builtin_amdgcn_wmma_f32_16x16x32_f16(false, a.v, false, b0.v,
                                                  (short)0, acc0, false, false);
    acc1 = __builtin_amdgcn_wmma_f32_16x16x32_f16(false, a.v, false, b1.v,
                                                  (short)0, acc1, false, false);
    acc2 = __builtin_amdgcn_wmma_f32_16x16x32_f16(false, a.v, false, b2.v,
                                                  (short)0, acc2, false, false);
    acc3 = __builtin_amdgcn_wmma_f32_16x16x32_f16(false, a.v, false, b3.v,
                                                  (short)0, acc3, false, false);
    ap += 32; bp += 32;
  }

  #pragma unroll
  for (int j = 0; j < 4; ++j) {
    v8f acc = (j == 0) ? acc0 : (j == 1) ? acc1 : (j == 2) ? acc2 : acc3;
    const int colg = (nt0 + j) * 16 + l15;
    const float bv = bias ? bias[colg] : 0.0f;
    #pragma unroll
    for (int r = 0; r < 8; ++r) {
      int rowg = mt * 16 + r + half * 8;
      float v = acc[r] + bv;
      if (do_relu) v = v > 0.0f ? v : 0.0f;
      C[(size_t)rowg * cols + colg] = (_Float16)v;
    }
  }
}

// ---------------------------------------------------------------------------
// Small glue kernels (tiny FLOP count)
// ---------------------------------------------------------------------------
__global__ void att_kernel(const _Float16* __restrict__ a1,
                           const float* __restrict__ w1,
                           const float* __restrict__ b1,
                           float* __restrict__ att, int rows) {
  int r = blockIdx.x * blockDim.x + threadIdx.x;
  if (r >= rows) return;
  const _Float16* ar = a1 + (size_t)r * HP;
  float s = 0.0f;
  for (int k = 0; k < HP; ++k) s += (float)ar[k] * w1[k];
  att[r] = s + b1[0];
}

__global__ void softmax_sel_kernel(const float* __restrict__ att,
                                   float* __restrict__ sel) {
  __shared__ float red[256];
  int n = blockIdx.x, t = threadIdx.x;
  float v = (t < OO) ? att[(size_t)n * OO + t] : -1e30f;
  red[t] = v; __syncthreads();
  for (int s = 128; s > 0; s >>= 1) {
    if (t < s) red[t] = fmaxf(red[t], red[t + s]);
    __syncthreads();
  }
  float m = red[0]; __syncthreads();
  float e = (t < OO) ? __expf(v - m) : 0.0f;
  red[t] = e; __syncthreads();
  for (int s = 128; s > 0; s >>= 1) {
    if (t < s) red[t] += red[t + s];
    __syncthreads();
  }
  float inv = 1.0f / red[0];
  if (t < OO) sel[(size_t)n * OO + t] = e * inv;
}

__global__ void selected_kernel(const float* __restrict__ sel,
                                const _Float16* __restrict__ qe,
                                _Float16* __restrict__ selh) {   // [n][130]
  int n = blockIdx.x, d = threadIdx.x;  // 160 threads
  if (d >= CF + 2) return;
  float s = 0.0f;
  for (int o = 0; o < OO; ++o)
    s += sel[(size_t)n * OO + o] * (float)qe[((size_t)n * OO + o) * DQ + d];
  selh[(size_t)n * (CF + 2) + d] = (_Float16)s;
}

__global__ void pairs_kernel(const _Float16* __restrict__ qe,
                             const _Float16* __restrict__ selh,
                             _Float16* __restrict__ pairs) {     // [row][DP]
  int row = blockIdx.x;
  int n = row / OO;
  const _Float16* q = qe + (size_t)row * DQ;
  const _Float16* s = selh + (size_t)n * (CF + 2);
  _Float16* pr = pairs + (size_t)row * DP;
  for (int j = threadIdx.x; j < DP; j += blockDim.x) {
    _Float16 v;
    if (j < 258)      v = q[j];
    else if (j < 388) v = s[j - 258];
    else              v = (_Float16)0;
    pr[j] = v;
  }
}

__global__ void relations_kernel(const _Float16* __restrict__ g2,
                                 _Float16* __restrict__ relh) {  // [n][GT]
  int n = blockIdx.x, c = threadIdx.x;  // 256 threads
  float s = 0.0f;
  for (int o = 0; o < OO; ++o)
    s += (float)g2[((size_t)n * OO + o) * GT + c];
  relh[(size_t)n * GT + c] = (_Float16)s;
}

__global__ void final_kernel(const _Float16* __restrict__ f1,
                             const float* __restrict__ w,  // [256][32]
                             const float* __restrict__ b,
                             float* __restrict__ out) {    // [n][32]
  int n = blockIdx.x, a = threadIdx.x;  // 32 threads
  float s = 0.0f;
  for (int k = 0; k < FP; ++k) s += (float)f1[(size_t)n * FP + k] * w[k * AA + a];
  out[(size_t)n * AA + a] = s + b[a];
}

// ---------------------------------------------------------------------------
extern "C" void kernel_launch(void* const* d_in, const int* in_sizes, int n_in,
                              void* d_out, int out_size, void* d_ws, size_t ws_size,
                              hipStream_t stream) {
  (void)in_sizes; (void)n_in; (void)out_size; (void)ws_size;
  const float* image  = (const float*)d_in[0];
  const float* code   = (const float*)d_in[1];
  const float* conv_w = (const float*)d_in[2];
  const float* conv_b = (const float*)d_in[3];
  const float* hp_w0  = (const float*)d_in[4];
  const float* hp_b0  = (const float*)d_in[5];
  const float* hp_w1  = (const float*)d_in[6];
  const float* hp_b1  = (const float*)d_in[7];
  const float* gt_w0  = (const float*)d_in[8];
  const float* gt_b0  = (const float*)d_in[9];
  const float* gt_w1  = (const float*)d_in[10];
  const float* gt_b1  = (const float*)d_in[11];
  const float* fp_w0  = (const float*)d_in[12];
  const float* fp_b0  = (const float*)d_in[13];
  const float* fp_w1  = (const float*)d_in[14];
  const float* fp_b1  = (const float*)d_in[15];
  float* out = (float*)d_out;

  // ---- workspace carve-up (256B aligned), with aliasing ----
  char* ws = (char*)d_ws;
  size_t off = 0;
  auto carve = [&](size_t bytes) -> char* {
    char* p = ws + off;
    off += (bytes + 255) & ~(size_t)255;
    return p;
  };
  const size_t imgT_bytes  = (size_t)NB * OO * CIN * 2;  // 51.4 MB (dead after conv)
  char* region_img = carve(imgT_bytes);
  _Float16* imgT  = (_Float16*)region_img;
  // alias: pairs at region_img+0 (20.9 MB), g2 after it (12.9 MB)
  _Float16* pairs = (_Float16*)region_img;
  _Float16* g2    = (_Float16*)(region_img + ((((size_t)ROWS * DP * 2) + 4095) & ~(size_t)4095));

  _Float16* wt9    = (_Float16*)carve((size_t)9 * CF * CIN * 2);
  _Float16* qe     = (_Float16*)carve((size_t)ROWS * DQ * 2);
  _Float16* hp_w0T = (_Float16*)carve((size_t)HP * DQ * 2);
  _Float16* gt_w0T = (_Float16*)carve((size_t)GT * DP * 2);
  _Float16* gt_w1T = (_Float16*)carve((size_t)GT * GT * 2);
  _Float16* fp_w0T = (_Float16*)carve((size_t)FP * GT * 2);
  _Float16* a1     = (_Float16*)carve((size_t)ROWS * HP * 2);  // also g1 later
  _Float16* g1     = a1;
  float*    att    = (float*)carve((size_t)ROWS * 4);
  float*    sel    = (float*)carve((size_t)NB * OO * 4);
  _Float16* selh   = (_Float16*)carve((size_t)NB * (CF + 2) * 2);
  _Float16* relh   = (_Float16*)carve((size_t)NB * GT * 2);
  _Float16* f1h    = (_Float16*)carve((size_t)NB * FP * 2);

  // ---- 1. conversions / weight prep ----
  {
    size_t total = (size_t)NB * OO * CIN;
    img_convert_kernel<<<(unsigned)((total + 255) / 256), 256, 0, stream>>>(image, imgT);
  }
  wt_convert_kernel<<<(9 * CF * CIN + 255) / 256, 256, 0, stream>>>(conv_w, wt9);
  w_transpose_kernel<<<(HP * DQ + 255) / 256, 256, 0, stream>>>(hp_w0, hp_w0T, 258, DQ, HP);
  w_transpose_kernel<<<(GT * DP + 255) / 256, 256, 0, stream>>>(gt_w0, gt_w0T, 388, DP, GT);
  w_transpose_kernel<<<(GT * GT + 255) / 256, 256, 0, stream>>>(gt_w1, gt_w1T, GT, GT, GT);
  w_transpose_kernel<<<(FP * GT + 255) / 256, 256, 0, stream>>>(fp_w0, fp_w0T, GT, GT, FP);
  assemble_qe_kernel<<<ROWS, 160, 0, stream>>>(code, qe);

  // ---- 2. conv (implicit GEMM, WMMA, 64x16 tiles/wave) ----
  conv_wmma_kernel<<<dim3(CF / 64, 13, NB), 32, 0, stream>>>(imgT, wt9, conv_b, qe);

  // ---- 3. h_psi layer 0: a1 = relu(qe @ hp_w0 + b0) ----
  gemm_wmma_kernel<<<dim3(ROWS / 16, HP / 64), 32, 0, stream>>>(qe, DQ, hp_w0T, hp_b0, a1, HP, 1);

  // ---- 4. attention logits + softmax + soft selection ----
  att_kernel<<<(ROWS + 255) / 256, 256, 0, stream>>>(a1, hp_w1, hp_b1, att, ROWS);
  softmax_sel_kernel<<<NB, 256, 0, stream>>>(att, sel);
  selected_kernel<<<NB, 160, 0, stream>>>(sel, qe, selh);

  // ---- 5. build pairs (aliases imgT region; imgT is dead) ----
  pairs_kernel<<<ROWS, 128, 0, stream>>>(qe, selh, pairs);

  // ---- 6. g_theta: g1 = relu(pairs @ gt_w0 + b0); g2 = relu(g1 @ gt_w1 + b1) ----
  gemm_wmma_kernel<<<dim3(ROWS / 16, GT / 64), 32, 0, stream>>>(pairs, DP, gt_w0T, gt_b0, g1, GT, 1);
  gemm_wmma_kernel<<<dim3(ROWS / 16, GT / 64), 32, 0, stream>>>(g1, GT, gt_w1T, gt_b1, g2, GT, 1);

  // ---- 7. sum over objects ----
  relations_kernel<<<NB, 256, 0, stream>>>(g2, relh);

  // ---- 8. f_phi ----
  gemm_wmma_kernel<<<dim3(NB / 16, FP / 64), 32, 0, stream>>>(relh, GT, fp_w0T, fp_b0, f1h, FP, 1);
  final_kernel<<<NB, AA, 0, stream>>>(f1h, fp_w1, fp_b1, out);
}
